// RWKVSelfAttention_69818988364544
// MI455X (gfx1250) — matrix operand order repaired and checked by reading, not compiled, not run on hardware
//
#include <hip/hip_runtime.h>
#include <hip/hip_bf16.h>

// ---------------------------------------------------------------------------
// RWKV self-attention for MI455X (gfx1250, wave32, WMMA).
// B=4, T=4096, H=1024. Four [16384,1024]x[1024,1024] GEMMs on the bf16 WMMA
// path with f32 accumulation (symmetric double-buffered fragment pipeline);
// chunked (3-pass) WKV scan for parallelism.
// ---------------------------------------------------------------------------

typedef __attribute__((ext_vector_type(16))) __bf16 v16bf;
typedef __attribute__((ext_vector_type(8)))  __bf16 bf16x8;
typedef __attribute__((ext_vector_type(8)))  float  v8f;

#if defined(__has_builtin)
#if __has_builtin(__builtin_amdgcn_sched_barrier)
#define SCHED_FENCE() __builtin_amdgcn_sched_barrier(0)
#endif
#endif
#ifndef SCHED_FENCE
#define SCHED_FENCE()
#endif

#define Bdim 4
#define Tdim 4096
#define Hdim 1024
#define BT   (Bdim * Tdim)          // 16384 rows
#define NCH  32                     // scan chunks
#define CHL  (Tdim / NCH)           // 128 steps per chunk

// --------------------------------------------------------------------------
// xs = x + time_shift(x);  xk/xv/xr = bf16(xs * time_mix_*)
// --------------------------------------------------------------------------
__global__ void mix_kernel(const float* __restrict__ x,
                           const float* __restrict__ tmk,
                           const float* __restrict__ tmv,
                           const float* __restrict__ tmr,
                           __bf16* __restrict__ xk,
                           __bf16* __restrict__ xv,
                           __bf16* __restrict__ xr) {
  size_t g = (size_t)blockIdx.x * blockDim.x + threadIdx.x;   // < B*T*H
  int h = (int)(g & (Hdim - 1));
  int t = (int)((g >> 10) & (Tdim - 1));
  float xs = x[g] + (t > 0 ? x[g - Hdim] : 0.0f);
  xk[g] = (__bf16)(xs * tmk[h]);
  xv[g] = (__bf16)(xs * tmv[h]);
  xr[g] = (__bf16)(xs * tmr[h]);
}

// --------------------------------------------------------------------------
// fp32 -> bf16 weight conversion
// --------------------------------------------------------------------------
__global__ void cvt_bf16(const float* __restrict__ src, __bf16* __restrict__ dst) {
  size_t g = (size_t)blockIdx.x * blockDim.x + threadIdx.x;
  dst[g] = (__bf16)src[g];
}

// --------------------------------------------------------------------------
// WMMA fragment helpers
// --------------------------------------------------------------------------
struct Frag {
  union {
    v16bf  v;
    bf16x8 h[2];
  };
};

// A 16x32 bf16 layout: lanes 0-15 rows, vgpr0-3 = K[kb..kb+7],
// vgpr4-7 = K[kb+16..kb+23]; lanes 16-31 use kb += 8 (folded into base ptr).
// B 32x16 bf16 layout: lane = column, vgpr0-3 = K[kb..kb+7],
// vgpr4-7 = K[kb+16..kb+23]; lanes 16-31 use kb += 8 (folded into base ptr).
__device__ __forceinline__ void load_frags(const __bf16* const (&apB)[4],
                                           const __bf16* const (&wpB)[2],
                                           int kk, Frag (&af)[4], Frag (&bf)[2]) {
#pragma unroll
  for (int i = 0; i < 4; ++i) {
    const __bf16* ap = apB[i] + kk;
    af[i].h[0] = *(const bf16x8*)(ap);
    af[i].h[1] = *(const bf16x8*)(ap + 16);
  }
#pragma unroll
  for (int j = 0; j < 2; ++j) {
    const __bf16* wp = wpB[j] + kk;
    bf[j].h[0] = *(const bf16x8*)(wp);
    bf[j].h[1] = *(const bf16x8*)(wp + 16);
  }
}

__device__ __forceinline__ void mma_tile(const Frag (&af)[4], const Frag (&bf)[2],
                                         v8f (&acc)[4][2]) {
#pragma unroll
  for (int i = 0; i < 4; ++i)
#pragma unroll
    for (int j = 0; j < 2; ++j)
      acc[i][j] = __builtin_amdgcn_wmma_f32_16x16x32_bf16(
          false, af[i].v, false, bf[j].v, (short)0, acc[i][j], false, false);
}

// --------------------------------------------------------------------------
// C[M,N] = A[M,K](bf16,row-major) * W[N,K](bf16,row-major)^T via WMMA.
// Block: 256 threads = 8 waves, 128x128 tile; wave tile 64x32 (4x2 wmma).
// Symmetric double buffering: each buffer's 12 loads issue one full compute
// block ahead of their consumer; sched_barrier(0) pins the pipeline shape.
// mode 0: store fp32.  mode 1: sigmoid -> bf16.  mode 2: +bias -> fp32.
// --------------------------------------------------------------------------
__global__ __launch_bounds__(256, 1) void gemm_bf16(
    const __bf16* __restrict__ A, const __bf16* __restrict__ W,
    float* __restrict__ outF, __bf16* __restrict__ outB,
    const float* __restrict__ bias, int M, int N, int K, int mode) {
  const int lane  = threadIdx.x & 31;
  const int wave  = threadIdx.x >> 5;
  const int lmod  = lane & 15;
  const int lhalf = lane >> 4;                 // 0 or 1
  const int waveM0 = blockIdx.y * 128 + (wave >> 2) * 64;
  const int waveN0 = blockIdx.x * 128 + (wave & 3) * 32;

  // Per-fragment base pointers (lane-half K offset folded in).
  const __bf16* apB[4];
  const __bf16* wpB[2];
#pragma unroll
  for (int i = 0; i < 4; ++i)
    apB[i] = A + (size_t)(waveM0 + i * 16 + lmod) * K + (lhalf << 3);
#pragma unroll
  for (int j = 0; j < 2; ++j)
    wpB[j] = W + (size_t)(waveN0 + j * 16 + lmod) * K + (lhalf << 3);

  v8f acc[4][2];
#pragma unroll
  for (int i = 0; i < 4; ++i)
#pragma unroll
    for (int j = 0; j < 2; ++j) {
      v8f z = {0.f, 0.f, 0.f, 0.f, 0.f, 0.f, 0.f, 0.f};
      acc[i][j] = z;
    }

  Frag a0[4], b0[2], a1[4], b1[2];
  load_frags(apB, wpB, 0, a0, b0);
  load_frags(apB, wpB, 32, a1, b1);
  SCHED_FENCE();

  // K multiple of 64: manual 2x unroll, both buffers kept one full compute
  // block ahead of their consumer (no dynamic register-array indexing).
  for (int kk = 0; kk < K; kk += 64) {
    if (kk + 192 < K) {                          // keep a far K-slice in GL2
      __builtin_prefetch((const void*)(apB[0] + kk + 192), 0, 3);
      __builtin_prefetch((const void*)(wpB[0] + kk + 192), 0, 3);
    }
    mma_tile(a0, b0, acc);                       // compute buffer 0
    if (kk + 64 < K)
      load_frags(apB, wpB, kk + 64, a0, b0);     // refill buffer 0
    SCHED_FENCE();
    mma_tile(a1, b1, acc);                       // compute buffer 1
    if (kk + 96 < K)
      load_frags(apB, wpB, kk + 96, a1, b1);     // refill buffer 1
    SCHED_FENCE();
  }

  // C/D layout: vgpr r -> row r (lanes 0-15) / row r+8 (lanes 16-31),
  // col = lane&15.
#pragma unroll
  for (int i = 0; i < 4; ++i) {
#pragma unroll
    for (int j = 0; j < 2; ++j) {
      const int col = waveN0 + j * 16 + lmod;
#pragma unroll
      for (int r = 0; r < 8; ++r) {
        const int row = waveM0 + i * 16 + r + lhalf * 8;
        const size_t o = (size_t)row * N + col;
        const float val = acc[i][j][r];
        if (mode == 0) {
          outF[o] = val;
        } else if (mode == 1) {
          outB[o] = (__bf16)(1.0f / (1.0f + __expf(-val)));
        } else {
          outF[o] = val + bias[col];
        }
      }
    }
  }
}

// --------------------------------------------------------------------------
// WKV scan, chunked.  Recurrences are affine with constant decay per h:
//   num' = td*num + e^k * v ;  den' = td*den + e^k
// Pass 1: per-chunk affine offsets (C, D).  Layout [b][ch][h] (coalesced).
// --------------------------------------------------------------------------
__global__ void wkv_pass1(const float* __restrict__ kf,
                          const float* __restrict__ vf,
                          const float* __restrict__ time_decay,
                          float* __restrict__ Cc, float* __restrict__ Dc) {
  size_t g = (size_t)blockIdx.x * blockDim.x + threadIdx.x;   // < B*NCH*H
  int h  = (int)(g & (Hdim - 1));
  int ch = (int)((g >> 10) & (NCH - 1));
  int b  = (int)(g >> 15);
  const float td = time_decay[h];
  size_t idx = ((size_t)b * Tdim + (size_t)ch * CHL) * Hdim + h;
  float c = 0.0f, d = 0.0f;
  for (int t = 0; t < CHL; ++t, idx += Hdim) {
    float ek = __expf(kf[idx]);
    c = td * c + ek * vf[idx];
    d = td * d + ek;
  }
  Cc[g] = c;
  Dc[g] = d;
}

// Pass 2: tiny sequential prefix over NCH chunks per (b,h):
//   start(ch) = td^CHL * start(ch-1) + {C,D}(ch-1),  start(0) = 1.
__global__ void wkv_pass2(const float* __restrict__ Cc,
                          const float* __restrict__ Dc,
                          const float* __restrict__ time_decay,
                          float* __restrict__ numS, float* __restrict__ denS) {
  size_t g = (size_t)blockIdx.x * blockDim.x + threadIdx.x;   // < B*H
  int h = (int)(g & (Hdim - 1));
  int b = (int)(g >> 10);
  const float td  = time_decay[h];
  const float tdL = __powf(td, (float)CHL);
  float n = 1.0f, d = 1.0f;
  for (int ch = 0; ch < NCH; ++ch) {
    size_t o = ((size_t)b * NCH + ch) * Hdim + h;
    numS[o] = n;
    denS[o] = d;
    n = tdL * n + Cc[o];
    d = tdL * d + Dc[o];
  }
}

// Pass 3: replay each chunk from its start state, emit rout = bf16(r * wkv).
__global__ void wkv_pass3(const float* __restrict__ kf,
                          const float* __restrict__ vf,
                          const __bf16* __restrict__ rb,
                          const float* __restrict__ time_decay,
                          const float* __restrict__ numS,
                          const float* __restrict__ denS,
                          __bf16* __restrict__ rout) {
  size_t g = (size_t)blockIdx.x * blockDim.x + threadIdx.x;   // < B*NCH*H
  int h  = (int)(g & (Hdim - 1));
  int ch = (int)((g >> 10) & (NCH - 1));
  int b  = (int)(g >> 15);
  const float td = time_decay[h];
  float num = numS[g];
  float den = denS[g];
  size_t idx = ((size_t)b * Tdim + (size_t)ch * CHL) * Hdim + h;
  for (int t = 0; t < CHL; ++t, idx += Hdim) {
    float ck = kf[idx], cv = vf[idx];
    float tdk = td * ck;
    float out = (num + tdk * den) / (den + tdk);
    float ek = __expf(ck);
    num = td * num + ek * cv;
    den = td * den + ek;
    rout[idx] = (__bf16)((float)rb[idx] * out);
  }
}

// --------------------------------------------------------------------------
extern "C" void kernel_launch(void* const* d_in, const int* in_sizes, int n_in,
                              void* d_out, int out_size, void* d_ws, size_t ws_size,
                              hipStream_t stream) {
  const float* x   = (const float*)d_in[0];
  const float* td  = (const float*)d_in[1];
  const float* tmk = (const float*)d_in[2];
  const float* tmv = (const float*)d_in[3];
  const float* tmr = (const float*)d_in[4];
  const float* Wr  = (const float*)d_in[5];
  const float* Wk  = (const float*)d_in[6];
  const float* Wv  = (const float*)d_in[7];
  const float* Wo  = (const float*)d_in[8];
  const float* bo  = (const float*)d_in[9];

  const size_t NE = (size_t)BT * Hdim;        // 16M activations
  const size_t HH = (size_t)Hdim * Hdim;      // 1M weight elements
  const size_t SC = (size_t)Bdim * NCH * Hdim;// 128K scan chunk states

  char* ws = (char*)d_ws;
  size_t off = 0;
  __bf16* xk   = (__bf16*)(ws + off); off += 2 * NE;
  __bf16* xv   = (__bf16*)(ws + off); off += 2 * NE;
  __bf16* xr   = (__bf16*)(ws + off); off += 2 * NE;
  __bf16* wkb  = (__bf16*)(ws + off); off += 2 * HH;
  __bf16* wvb  = (__bf16*)(ws + off); off += 2 * HH;
  __bf16* wrb  = (__bf16*)(ws + off); off += 2 * HH;
  __bf16* wob  = (__bf16*)(ws + off); off += 2 * HH;
  float*  kf   = (float*)(ws + off);  off += 4 * NE;
  float*  vf   = (float*)(ws + off);  off += 4 * NE;
  __bf16* rbuf = (__bf16*)(ws + off); off += 2 * NE;
  float*  Cc   = (float*)(ws + off);  off += 4 * SC;
  float*  Dc   = (float*)(ws + off);  off += 4 * SC;
  float*  numS = (float*)(ws + off);  off += 4 * SC;
  float*  denS = (float*)(ws + off);  off += 4 * SC;
  __bf16* rout = xk;  // xk is dead after the k-GEMM; reuse its space

  // 1) time-shift + mix + bf16 downconvert
  mix_kernel<<<(unsigned)(NE / 256), 256, 0, stream>>>(x, tmk, tmv, tmr, xk, xv, xr);

  // 2) weights -> bf16
  cvt_bf16<<<(unsigned)(HH / 256), 256, 0, stream>>>(Wk, wkb);
  cvt_bf16<<<(unsigned)(HH / 256), 256, 0, stream>>>(Wv, wvb);
  cvt_bf16<<<(unsigned)(HH / 256), 256, 0, stream>>>(Wr, wrb);
  cvt_bf16<<<(unsigned)(HH / 256), 256, 0, stream>>>(Wo, wob);

  // 3) projection GEMMs on the WMMA path
  dim3 ggrid(Hdim / 128, BT / 128);   // (8, 128)
  gemm_bf16<<<ggrid, 256, 0, stream>>>(xk, wkb, kf, nullptr, nullptr,
                                       BT, Hdim, Hdim, 0);
  gemm_bf16<<<ggrid, 256, 0, stream>>>(xv, wvb, vf, nullptr, nullptr,
                                       BT, Hdim, Hdim, 0);
  gemm_bf16<<<ggrid, 256, 0, stream>>>(xr, wrb, nullptr, rbuf, nullptr,
                                       BT, Hdim, Hdim, 1);

  // 4) chunked WKV scan
  wkv_pass1<<<(unsigned)(SC / 256), 256, 0, stream>>>(kf, vf, td, Cc, Dc);
  wkv_pass2<<<(unsigned)((size_t)Bdim * Hdim / 256), 256, 0, stream>>>(
      Cc, Dc, td, numS, denS);
  wkv_pass3<<<(unsigned)(SC / 256), 256, 0, stream>>>(kf, vf, rbuf, td,
                                                      numS, denS, rout);

  // 5) output projection + bias -> fp32 d_out
  gemm_bf16<<<ggrid, 256, 0, stream>>>(rout, wob, (float*)d_out, nullptr, bo,
                                       BT, Hdim, Hdim, 2);
}
